// SASRec_38912403702077
// MI455X (gfx1250) — matrix operand order repaired and checked by reading, not compiled, not run on hardware
//
#include <hip/hip_runtime.h>
#include <hip/hip_bf16.h>

// ---------------------------------------------------------------------------
// MI455X / gfx1250, wave32. WMMA f32<-bf16 16x16x32.
// Bandwidth-bound workload (~28 GFLOP vs ~2 GB irregular reads @ 23.3 TB/s).
// Weights are pre-packed into the WMMA B-fragment layout (bf16, zero-padded)
// so the GEMM inner loop is: 1x 32B global load (B) + 2x ds_load_b128 (A)
// + 1x v_wmma per chunk -- no per-element masking, no serializing waits.
// ---------------------------------------------------------------------------

typedef __attribute__((ext_vector_type(16))) __bf16 v16bf;
typedef __attribute__((ext_vector_type(8)))  float  v8f;

__device__ __forceinline__ v8f wmma_bf16(v16bf a, v16bf b, v8f c) {
    // (neg_a, A, neg_b, B, c_mod, C, reuse_a, reuse_b)
    return __builtin_amdgcn_wmma_f32_16x16x32_bf16(false, a, false, b,
                                                   (short)0, c, false, false);
}

// A-fragment (16x32 bf16, ISA 7.12.2): row = lane&15, half = lane>>4 selects
// the K sub-stripe. Reads are two contiguous 16B runs -> 2x ds_load_b128.
__device__ __forceinline__ v16bf load_A_frag(const __bf16* __restrict__ rowp,
                                             int k0, int half) {
    v16bf a;
#pragma unroll
    for (int v = 0; v < 8; ++v) {
        const int koff = k0 + ((v & 3) << 1) + (half << 3) + ((v >> 2) << 4);
        a[2 * v]     = rowp[koff];
        a[2 * v + 1] = rowp[koff + 1];
    }
    return a;
}

// Packed-weight layout: [kchunk][coltile(4)][lane(32)][16 bf16]
// -> each lane's 16 B-fragment values are 32 contiguous bytes.
__device__ __forceinline__ v16bf load_B_packed(const __bf16* __restrict__ Wp,
                                               int kc, int wv, int lane) {
    return *(const v16bf*)(Wp + ((((kc * 4 + wv) * 32) + lane) << 4));
}

// ---------------------------------------------------------------------------
// Pack fp32 [Kreal x 64] weight into bf16 B-fragment layout, zero-padded to
// Kpad rows. out has Kpad*64 bf16 elements.
//   idx = ((kc*4 + ct)*32 + lane)*16 + j
//   k   = kc*32 + (lane>>4)*8 + (j&7) + (j>>3)*16 ;  col = ct*16 + (lane&15)
// ---------------------------------------------------------------------------
__global__ __launch_bounds__(256) void pack_weights_kernel(
    const float* __restrict__ W, __bf16* __restrict__ out, int Kreal, int Kpad)
{
    const int total = Kpad * 64;
    for (int idx = blockIdx.x * blockDim.x + threadIdx.x; idx < total;
         idx += gridDim.x * blockDim.x) {
        const int j    = idx & 15;
        const int lane = (idx >> 4) & 31;
        const int ct   = (idx >> 9) & 3;
        const int kc   = idx >> 11;
        const int k    = kc * 32 + ((lane >> 4) << 3) + (j & 7) + ((j >> 3) << 4);
        const int col  = ct * 16 + (lane & 15);
        const float f  = (k < Kreal) ? W[k * 64 + col] : 0.0f;
        out[idx] = (__bf16)f;
    }
}

// ---------------------------------------------------------------------------
// Phase 1: build concatenated input features in LDS (bf16) and project to 64.
//   users: [id 64 | feat-bag 64 | 3x word-mean 96 | word300]        Kreal=524
//   items: [ ... same ...                        | sentence768]     Kreal=1292
// Epilogue adds proj_b + numeric @ num_W + num_b.
// ---------------------------------------------------------------------------
__global__ __launch_bounds__(128) void feat_proj_kernel(
    const float*  __restrict__ id_emb,          // [n, 64]
    const float*  __restrict__ feat_table,      // [*, 64]
    const int*    __restrict__ feat_idx,        // [n, 5]
    const float*  __restrict__ word_emb,        // [30000, 32]
    const int*    __restrict__ text_idx,        // [n, 3, 8]
    const float*  __restrict__ word_embedding,  // [n, 300]  (streamed, NT)
    const float*  __restrict__ sentence_emb,    // [n, 768] or nullptr (NT)
    const float*  __restrict__ numeric,         // [n, 10]
    const float*  __restrict__ num_W,           // [10, 64]
    const float*  __restrict__ num_b,           // [64]
    const __bf16* __restrict__ projWp,          // packed [Kpad/32][4][32][16]
    const float*  __restrict__ proj_b,          // [64]
    float* __restrict__ xout,                   // [N, 64]
    int row_base, int Kreal, int Kpad)
{
    __shared__ __bf16 sIn[16 * 1312];           // max item tile: 41984 B
    const int tile = blockIdx.x * 16;

    __builtin_prefetch((const void*)projWp, 0, 1);   // global_prefetch_b8

    // ---- build 16 x Kpad bf16 input tile -------------------------------
    for (int r = 0; r < 16; ++r) {
        const int u = tile + r;
        __bf16* dst = sIn + r * Kpad;
        for (int c = threadIdx.x; c < Kpad; c += blockDim.x) {
            float v;
            if (c < 64) {
                v = id_emb[u * 64 + c];
            } else if (c < 128) {                    // EmbeddingBag mean (5)
                const int cc = c - 64;
                float s = 0.f;
#pragma unroll
                for (int k = 0; k < 5; ++k)
                    s += feat_table[feat_idx[u * 5 + k] * 64 + cc];
                v = s * 0.2f;
            } else if (c < 224) {                    // 3 text fields, mean(8)
                const int cc = c - 128;
                const int f = cc >> 5, c32 = cc & 31;
                float s = 0.f;
#pragma unroll
                for (int wd = 0; wd < 8; ++wd)
                    s += word_emb[text_idx[(u * 3 + f) * 8 + wd] * 32 + c32];
                v = s * 0.125f;
            } else if (c < 524) {
                v = __builtin_nontemporal_load(word_embedding + u * 300 + (c - 224));
            } else if (c < Kreal) {
                v = __builtin_nontemporal_load(sentence_emb + u * 768 + (c - 524));
            } else {
                v = 0.f;                             // zero pad to Kpad
            }
            dst[c] = (__bf16)v;
        }
    }
    __syncthreads();

    // ---- 4 waves x (16 rows x 16 cols) WMMA over K chunks --------------
    const int lane = threadIdx.x & 31;
    const int wv   = threadIdx.x >> 5;
    const int half = lane >> 4;
    const int mrow = lane & 15;
    const int col  = wv * 16 + mrow;

    v8f acc = {};
    const __bf16* rowp = sIn + mrow * Kpad;
    const int nchunks = Kpad >> 5;
    for (int kc = 0; kc < nchunks; ++kc) {
        v16bf a = load_A_frag(rowp, kc * 32, half);
        v16bf b = load_B_packed(projWp, kc, wv, lane);
        acc = wmma_bf16(a, b, acc);
    }

    // ---- epilogue: + proj_b + numeric@num_W + num_b --------------------
    const float bias = proj_b[col] + num_b[col];
    float nw[10];
#pragma unroll
    for (int t = 0; t < 10; ++t) nw[t] = num_W[t * 64 + col];
#pragma unroll
    for (int r = 0; r < 8; ++r) {
        const int row = r + 8 * half;
        const int u   = tile + row;
        float v = acc[r] + bias;
#pragma unroll
        for (int t = 0; t < 10; ++t) v += numeric[u * 10 + t] * nw[t];
        xout[(size_t)(row_base + u) * 64 + col] = v;
    }
}

// ---------------------------------------------------------------------------
// Phase 2: one GNN layer per 16-node tile, fully fused:
//   nbrMean (K=8 gather, float4) -> LDS bf16
//   agg  = nbrMean @ vW + vb                          (2 WMMA / wave)
//   xnew = self @ wW[0:64] + agg @ wW[64:128] + wb    (4 WMMA / wave)
//   optional ReLU; final layer: row L2-normalize, NT-store to d_out.
// ---------------------------------------------------------------------------
__global__ __launch_bounds__(128) void gnn_layer_kernel(
    const float*  __restrict__ xin,        // [N, 64]
    const int*    __restrict__ neighbors,  // [N, 8]
    const __bf16* __restrict__ vWp,        // packed, 2 kchunks
    const float*  __restrict__ vb,         // [64]
    const __bf16* __restrict__ wWp,        // packed, 4 kchunks
    const float*  __restrict__ wb,         // [64]
    float* __restrict__ xout,              // [N, 64]
    int do_relu, int do_norm)
{
    __shared__ __bf16 sSelf[16][64];
    __shared__ __bf16 sNbr[16][64];
    __shared__ __bf16 sAgg[16][64];
    __shared__ float  sOut[16][64];
    __shared__ float  sScale[16];

    const int tile = blockIdx.x * 16;
    __builtin_prefetch((const void*)wWp, 0, 1);

    // ---- stage 1: self row + neighbor mean, float4 coalesced -----------
    for (int t = threadIdx.x; t < 256; t += blockDim.x) {
        const int r = t >> 4, cg = (t & 15) << 2;
        const int n = tile + r;
        const float4 xs = *(const float4*)(xin + (size_t)n * 64 + cg);
        sSelf[r][cg + 0] = (__bf16)xs.x; sSelf[r][cg + 1] = (__bf16)xs.y;
        sSelf[r][cg + 2] = (__bf16)xs.z; sSelf[r][cg + 3] = (__bf16)xs.w;
        float ax = 0.f, ay = 0.f, az = 0.f, aw = 0.f;
#pragma unroll
        for (int k = 0; k < 8; ++k) {
            const int nb = neighbors[n * 8 + k];
            const float4 v = *(const float4*)(xin + (size_t)nb * 64 + cg);
            ax += v.x; ay += v.y; az += v.z; aw += v.w;
        }
        sNbr[r][cg + 0] = (__bf16)(ax * 0.125f);
        sNbr[r][cg + 1] = (__bf16)(ay * 0.125f);
        sNbr[r][cg + 2] = (__bf16)(az * 0.125f);
        sNbr[r][cg + 3] = (__bf16)(aw * 0.125f);
    }
    __syncthreads();

    const int lane = threadIdx.x & 31;
    const int wv   = threadIdx.x >> 5;
    const int half = lane >> 4;
    const int mrow = lane & 15;
    const int col  = wv * 16 + mrow;

    // ---- stage 2: agg = nbrMean @ vW + vb ------------------------------
    {
        v8f c = {};
#pragma unroll
        for (int kc = 0; kc < 2; ++kc) {
            v16bf a = load_A_frag(&sNbr[mrow][0], kc * 32, half);
            v16bf b = load_B_packed(vWp, kc, wv, lane);
            c = wmma_bf16(a, b, c);
        }
        const float bias = vb[col];
#pragma unroll
        for (int r = 0; r < 8; ++r)
            sAgg[r + 8 * half][col] = (__bf16)(c[r] + bias);
    }
    __syncthreads();

    // ---- stage 3: xnew = self @ wW[0:64] + agg @ wW[64:128] + wb -------
    v8f d = {};
#pragma unroll
    for (int kc = 0; kc < 2; ++kc) {
        v16bf a = load_A_frag(&sSelf[mrow][0], kc * 32, half);
        v16bf b = load_B_packed(wWp, kc, wv, lane);        // rows 0..63
        d = wmma_bf16(a, b, d);
    }
#pragma unroll
    for (int kc = 0; kc < 2; ++kc) {
        v16bf a = load_A_frag(&sAgg[mrow][0], kc * 32, half);
        v16bf b = load_B_packed(wWp, kc + 2, wv, lane);    // rows 64..127
        d = wmma_bf16(a, b, d);
    }

    const float bias = wb[col];
    if (!do_norm) {
#pragma unroll
        for (int r = 0; r < 8; ++r) {
            const int row = r + 8 * half;
            float v = d[r] + bias;
            if (do_relu) v = fmaxf(v, 0.f);
            xout[(size_t)(tile + row) * 64 + col] = v;   // re-read next layer: keep cached
        }
    } else {
#pragma unroll
        for (int r = 0; r < 8; ++r) {
            float v = d[r] + bias;
            if (do_relu) v = fmaxf(v, 0.f);
            sOut[r + 8 * half][col] = v;
        }
        __syncthreads();
        if (threadIdx.x < 16) {
            float s = 0.f;
            for (int c = 0; c < 64; ++c) { const float v = sOut[threadIdx.x][c]; s += v * v; }
            sScale[threadIdx.x] = 1.0f / fmaxf(sqrtf(s), 1e-12f);
        }
        __syncthreads();
        for (int t = threadIdx.x; t < 1024; t += blockDim.x) {
            const int r = t >> 6, c = t & 63;
            __builtin_nontemporal_store(sOut[r][c] * sScale[r],
                                        xout + (size_t)(tile + r) * 64 + c);
        }
    }
}

// ---------------------------------------------------------------------------
extern "C" void kernel_launch(void* const* d_in, const int* in_sizes, int n_in,
                              void* d_out, int out_size, void* d_ws, size_t ws_size,
                              hipStream_t stream) {
    // setup_inputs() dict order:
    const float* user_id_emb        = (const float*)d_in[0];
    const float* item_id_emb        = (const float*)d_in[1];
    const float* user_feat_table    = (const float*)d_in[2];
    const float* item_feat_table    = (const float*)d_in[3];
    const float* word_emb           = (const float*)d_in[4];
    const float* user_numeric       = (const float*)d_in[5];
    const float* item_numeric       = (const float*)d_in[6];
    const float* user_word_embed    = (const float*)d_in[7];
    const float* item_word_embed    = (const float*)d_in[8];
    const float* item_sentence_emb  = (const float*)d_in[9];
    const float* user_num_W         = (const float*)d_in[10];
    const float* user_num_b         = (const float*)d_in[11];
    const float* item_num_W         = (const float*)d_in[12];
    const float* item_num_b         = (const float*)d_in[13];
    const float* user_proj_W        = (const float*)d_in[14];
    const float* user_proj_b        = (const float*)d_in[15];
    const float* item_proj_W        = (const float*)d_in[16];
    const float* item_proj_b        = (const float*)d_in[17];
    const float* w_W                = (const float*)d_in[18];   // [2,128,64]
    const float* w_b                = (const float*)d_in[19];   // [2,64]
    const float* v_W                = (const float*)d_in[20];   // [2,64,64]
    const float* v_b                = (const float*)d_in[21];   // [2,64]
    const int*   user_feat_idx      = (const int*)d_in[22];
    const int*   item_feat_idx      = (const int*)d_in[23];
    const int*   user_text_idx      = (const int*)d_in[24];
    const int*   item_text_idx      = (const int*)d_in[25];
    const int*   neighbors          = (const int*)d_in[26];

    constexpr int N_USER = 150000, M_ITEM = 50000, N = N_USER + M_ITEM, D = 64;

    // workspace layout:
    //   x0, x1 : ping/pong node features (2 x N x 64 f32 = 102.4 MB)
    //   packed bf16 weights (~0.3 MB, 32B-aligned)
    float* x0 = (float*)d_ws;
    float* x1 = x0 + (size_t)N * D;
    __bf16* wsb    = (__bf16*)(x1 + (size_t)N * D);
    __bf16* userWp = wsb;                  // 17 chunks * 2048 = 34816
    __bf16* itemWp = userWp + 17 * 2048;   // 41 chunks
    __bf16* vWp0   = itemWp + 41 * 2048;   //  2 chunks
    __bf16* vWp1   = vWp0 + 2 * 2048;
    __bf16* wWp0   = vWp1 + 2 * 2048;      //  4 chunks
    __bf16* wWp1   = wWp0 + 4 * 2048;

    // ---- pack weights to WMMA B-fragment bf16 layout -------------------
    pack_weights_kernel<<<(544 * 64 + 255) / 256, 256, 0, stream>>>(
        user_proj_W, userWp, 524, 544);
    pack_weights_kernel<<<(1312 * 64 + 255) / 256, 256, 0, stream>>>(
        item_proj_W, itemWp, 1292, 1312);
    pack_weights_kernel<<<(64 * 64 + 255) / 256, 256, 0, stream>>>(
        v_W, vWp0, 64, 64);
    pack_weights_kernel<<<(64 * 64 + 255) / 256, 256, 0, stream>>>(
        v_W + 64 * 64, vWp1, 64, 64);
    pack_weights_kernel<<<(128 * 64 + 255) / 256, 256, 0, stream>>>(
        w_W, wWp0, 128, 128);
    pack_weights_kernel<<<(128 * 64 + 255) / 256, 256, 0, stream>>>(
        w_W + 128 * 64, wWp1, 128, 128);

    // ---- phase 1: node features + projection ---------------------------
    feat_proj_kernel<<<N_USER / 16, 128, 0, stream>>>(
        user_id_emb, user_feat_table, user_feat_idx, word_emb, user_text_idx,
        user_word_embed, nullptr, user_numeric, user_num_W, user_num_b,
        userWp, user_proj_b, x0, /*row_base=*/0, /*Kreal=*/524, /*Kpad=*/544);

    feat_proj_kernel<<<M_ITEM / 16, 128, 0, stream>>>(
        item_id_emb, item_feat_table, item_feat_idx, word_emb, item_text_idx,
        item_word_embed, item_sentence_emb, item_numeric, item_num_W, item_num_b,
        itemWp, item_proj_b, x0, /*row_base=*/N_USER, /*Kreal=*/1292, /*Kpad=*/1312);

    // ---- phase 2: GNN layers (layer 0: ReLU; layer 1: normalize) -------
    gnn_layer_kernel<<<N / 16, 128, 0, stream>>>(
        x0, neighbors, vWp0, v_b, wWp0, w_b, x1, /*relu=*/1, /*norm=*/0);

    gnn_layer_kernel<<<N / 16, 128, 0, stream>>>(
        x1, neighbors, vWp1, v_b + 64, wWp1, w_b + 64,
        (float*)d_out, /*relu=*/0, /*norm=*/1);
}
